// GATGraphRegressor_46677704572988
// MI455X (gfx1250) — compile-verified, matrix operand order checked by reading, not run on hardware
//
#include <hip/hip_runtime.h>
#include <hip/hip_bf16.h>

typedef __attribute__((ext_vector_type(16))) _Float16 v16h;
typedef __attribute__((ext_vector_type(8)))  float    v8f;
typedef __attribute__((ext_vector_type(4)))  int      v4i;

#define NN 20000
#define EE 640000
#define HH 256
#define GG 256
#define ENED (EE + NN)
#define NEG_SLOPE 0.2f
#define BN_EPS 1e-5f

#define HAS_ASYNC_LDS __has_builtin(__builtin_amdgcn_global_load_async_to_lds_b128)

#if HAS_ASYNC_LDS
typedef __attribute__((address_space(1))) v4i* gas_p;   // global int4*
typedef __attribute__((address_space(3))) v4i* las_p;   // LDS int4*
#endif

// ---------------- helpers ----------------

__device__ __forceinline__ unsigned int mapOrdered(float v) {
    unsigned int u = __float_as_uint(v);
    return (u & 0x80000000u) ? ~u : (u | 0x80000000u);
}
__device__ __forceinline__ float unmapOrdered(unsigned int u) {
    unsigned int b = (u & 0x80000000u) ? (u & 0x7FFFFFFFu) : ~u;
    return __uint_as_float(b);
}

// ---------------- prep kernels ----------------

__global__ void k_cvt_f16(const float* __restrict__ src, _Float16* __restrict__ dst, int n) {
    int i = blockIdx.x * blockDim.x + threadIdx.x;
    if (i < n) dst[i] = (_Float16)src[i];
}

// W is [K, HH] row-major f32 -> WT [HH, K] row-major f16
__global__ void k_wtrans(const float* __restrict__ W, _Float16* __restrict__ WT, int K) {
    int i = blockIdx.x * blockDim.x + threadIdx.x;
    if (i >= K * HH) return;
    int k = i / HH;
    int h = i - k * HH;
    WT[(size_t)h * K + k] = (_Float16)W[(size_t)k * HH + h];
}

__global__ void k_fill_f32(float* __restrict__ p, float v, int n) {
    int i = blockIdx.x * blockDim.x + threadIdx.x;
    if (i < n) p[i] = v;
}
__global__ void k_fill_u32(unsigned int* __restrict__ p, unsigned int v, int n) {
    int i = blockIdx.x * blockDim.x + threadIdx.x;
    if (i < n) p[i] = v;
}

// ---------------- WMMA GEMM: C[NN,HH] = A[NN,K](f16) * BT[HH,K](f16)^T ----------------
// Block (256 thr, 8 waves) owns a 32-row M band. The 32xK f16 A panel is staged into LDS
// via async global->LDS loads (ASYNCcnt), then each wave computes a 16x64 output strip:
// one A fragment (from LDS, ds_load_b128) reused across 4 B fragments -> 4 WMMAs / K-step.
// 16-bit A/B VGPR layout: lane<16 holds K = 0..7 & 16..23; lane>=16 holds K = 8..15 & 24..31.

__global__ void __launch_bounds__(256) k_gemm_wmma(const _Float16* __restrict__ A,
                                                   const _Float16* __restrict__ BT,
                                                   float* __restrict__ C, int K) {
    __shared__ _Float16 Alds[32 * 256];   // up to 16 KB (K<=256)

    int tid  = threadIdx.x;
    int wave = tid >> 5;
    int lane = tid & 31;
    int r    = lane & 15;
    int half = lane >> 4;
    int klo  = half * 8;

    int blockRow = blockIdx.x * 32;       // 625 blocks, NN = 625*32
    int tm2 = wave >> 2;                  // 0..1 : which 16-row tile of the band
    int tg  = wave & 3;                   // 0..3 : N tile group (4 tiles of 16 cols)

#if HAS_ASYNC_LDS
    {
        const char* gsrc = (const char*)(A + (size_t)blockRow * K);
        char*       ldst = (char*)Alds;
        int bytes = 32 * K * (int)sizeof(_Float16);
        for (int o = tid * 16; o < bytes; o += 256 * 16) {
            __builtin_amdgcn_global_load_async_to_lds_b128(
                (gas_p)(gsrc + o), (las_p)(ldst + o), 0, 0);
        }
        asm volatile("s_wait_asynccnt 0x0" ::: "memory");
    }
    __syncthreads();
#endif

    int rowLocal = tm2 * 16 + r;
    const _Float16* brow = BT + (size_t)(tg * 64 + r) * K + klo;

    v8f acc0 = {}, acc1 = {}, acc2 = {}, acc3 = {};
    for (int k0 = 0; k0 < K; k0 += 32) {
        v16h a, b0, b1, b2, b3;
#if HAS_ASYNC_LDS
        {
            const _Float16* ap = Alds + rowLocal * K + klo + k0;
            *reinterpret_cast<uint4*>(&a)       = *reinterpret_cast<const uint4*>(ap);
            *(reinterpret_cast<uint4*>(&a) + 1) = *reinterpret_cast<const uint4*>(ap + 16);
        }
#else
        {
            const _Float16* ap = A + (size_t)(blockRow + rowLocal) * K + klo + k0;
            *reinterpret_cast<uint4*>(&a)       = *reinterpret_cast<const uint4*>(ap);
            *(reinterpret_cast<uint4*>(&a) + 1) = *reinterpret_cast<const uint4*>(ap + 16);
        }
#endif
        const _Float16* bp0 = brow + k0;
        const _Float16* bp1 = bp0 + (size_t)16 * K;
        const _Float16* bp2 = bp0 + (size_t)32 * K;
        const _Float16* bp3 = bp0 + (size_t)48 * K;
        *reinterpret_cast<uint4*>(&b0)       = *reinterpret_cast<const uint4*>(bp0);
        *(reinterpret_cast<uint4*>(&b0) + 1) = *reinterpret_cast<const uint4*>(bp0 + 16);
        *reinterpret_cast<uint4*>(&b1)       = *reinterpret_cast<const uint4*>(bp1);
        *(reinterpret_cast<uint4*>(&b1) + 1) = *reinterpret_cast<const uint4*>(bp1 + 16);
        *reinterpret_cast<uint4*>(&b2)       = *reinterpret_cast<const uint4*>(bp2);
        *(reinterpret_cast<uint4*>(&b2) + 1) = *reinterpret_cast<const uint4*>(bp2 + 16);
        *reinterpret_cast<uint4*>(&b3)       = *reinterpret_cast<const uint4*>(bp3);
        *(reinterpret_cast<uint4*>(&b3) + 1) = *reinterpret_cast<const uint4*>(bp3 + 16);

        acc0 = __builtin_amdgcn_wmma_f32_16x16x32_f16(false, a, false, b0, (short)0, acc0, false, false);
        acc1 = __builtin_amdgcn_wmma_f32_16x16x32_f16(false, a, false, b1, (short)0, acc1, false, false);
        acc2 = __builtin_amdgcn_wmma_f32_16x16x32_f16(false, a, false, b2, (short)0, acc2, false, false);
        acc3 = __builtin_amdgcn_wmma_f32_16x16x32_f16(false, a, false, b3, (short)0, acc3, false, false);
    }

    // D layout: VGPR v -> row M = v + 8*half, col N = r
    float* crow = C + (size_t)(blockRow + tm2 * 16 + half * 8) * HH + tg * 64 + r;
#pragma unroll
    for (int v = 0; v < 8; v++) {
        crow[(size_t)v * HH +  0] = acc0[v];
        crow[(size_t)v * HH + 16] = acc1[v];
        crow[(size_t)v * HH + 32] = acc2[v];
        crow[(size_t)v * HH + 48] = acc3[v];
    }
}

// ---------------- attention dot products ----------------

__global__ void k_dots(const float* __restrict__ h,
                       const float* __restrict__ asrc, const float* __restrict__ adst,
                       float* __restrict__ es, float* __restrict__ ed) {
    int gt   = blockIdx.x * blockDim.x + threadIdx.x;
    int node = gt >> 5;
    int lane = threadIdx.x & 31;
    if (node >= NN) return;
    const float* hr = h + (size_t)node * HH;
    float s = 0.f, d = 0.f;
#pragma unroll
    for (int t = 0; t < 8; t++) {
        int f = lane + 32 * t;
        float v = hr[f];
        s += v * asrc[f];
        d += v * adst[f];
    }
    for (int off = 16; off; off >>= 1) {
        s += __shfl_down(s, off, 32);
        d += __shfl_down(d, off, 32);
    }
    if (lane == 0) { es[node] = s; ed[node] = d; }
}

// ---------------- edge pass 1: leaky-relu logits + segment max ----------------

__global__ void k_edge_logit_max(const float* __restrict__ es, const float* __restrict__ ed,
                                 const int* __restrict__ src, const int* __restrict__ dst,
                                 float* __restrict__ L, unsigned int* __restrict__ mM) {
    int e = blockIdx.x * blockDim.x + threadIdx.x;
    if (e >= ENED) return;
    int s, d;
    if (e < EE) { s = src[e]; d = dst[e]; }
    else        { s = e - EE; d = s; }
    float v = es[s] + ed[d];
    v = (v > 0.f) ? v : NEG_SLOPE * v;
    L[e] = v;
    atomicMax(mM + d, mapOrdered(v));
}

// ---------------- edge pass 2: exp + segment sum ----------------

__global__ void k_edge_exp(const float* __restrict__ L, const unsigned int* __restrict__ mM,
                           const int* __restrict__ dst,
                           float* __restrict__ EX, float* __restrict__ z) {
    int e = blockIdx.x * blockDim.x + threadIdx.x;
    if (e >= ENED) return;
    int d = (e < EE) ? dst[e] : (e - EE);
    float ex = __expf(L[e] - unmapOrdered(mM[d]));
    EX[e] = ex;
    atomicAdd(z + d, ex);
}

// ---------------- edge pass 3: aggregate alpha * h[src] into agg[dst] ----------------

__global__ void k_edge_agg(const float* __restrict__ EX, const float* __restrict__ z,
                           const int* __restrict__ src, const int* __restrict__ dst,
                           const float* __restrict__ h, float* __restrict__ agg) {
    int gt   = blockIdx.x * blockDim.x + threadIdx.x;
    int e    = gt >> 5;
    int lane = threadIdx.x & 31;
    if (e >= ENED) return;
    int s, d;
    if (e < EE) { s = src[e]; d = dst[e]; }
    else        { s = e - EE; d = s; }
    float alpha = EX[e] / z[d];
    const float* hs = h + (size_t)s * HH;
    float* ad = agg + (size_t)d * HH;
    __builtin_prefetch(hs, 0, 0);   // global_prefetch_b8 on the gathered row
#pragma unroll
    for (int t = 0; t < 8; t++) {
        int f = lane + 32 * t;
        atomicAdd(ad + f, alpha * hs[f]);
    }
}

// ---------------- epilogue: +bias, BN, ReLU ----------------

__global__ void k_bn_relu(float* __restrict__ agg, const float* __restrict__ b,
                          const float* __restrict__ gam, const float* __restrict__ bet,
                          const float* __restrict__ mu, const float* __restrict__ var,
                          _Float16* __restrict__ xh) {
    int i = blockIdx.x * blockDim.x + threadIdx.x;
    if (i >= NN * HH) return;
    int f = i & (HH - 1);
    float v = agg[i] + b[f];
    v = (v - mu[f]) * rsqrtf(var[f] + BN_EPS) * gam[f] + bet[f];
    v = (v > 0.f) ? v : 0.f;
    agg[i] = v;
    xh[i]  = (_Float16)v;
}

// ---------------- graph mean pool + FC ----------------

__global__ void k_pool(const float* __restrict__ act, const int* __restrict__ batch,
                       float* __restrict__ psum, float* __restrict__ pcnt) {
    int i = blockIdx.x * blockDim.x + threadIdx.x;
    if (i >= NN * HH) return;
    int n = i >> 8;
    int f = i & 255;
    int g = batch[n];
    atomicAdd(psum + (size_t)g * HH + f, act[i]);
    if (f == 0) atomicAdd(pcnt + g, 1.0f);
}

__global__ void k_fc(const float* __restrict__ psum, const float* __restrict__ pcnt,
                     const float* __restrict__ fcW, const float* __restrict__ fcb,
                     float* __restrict__ out) {
    int gt   = blockIdx.x * blockDim.x + threadIdx.x;
    int g    = gt >> 5;
    int lane = threadIdx.x & 31;
    if (g >= GG) return;
    float c = pcnt[g];
    c = (c > 1.f) ? c : 1.f;
    float acc = 0.f;
#pragma unroll
    for (int t = 0; t < 8; t++) {
        int f = lane + 32 * t;
        acc += (psum[(size_t)g * HH + f] / c) * fcW[f];
    }
    for (int off = 16; off; off >>= 1) acc += __shfl_down(acc, off, 32);
    if (lane == 0) out[g] = acc + fcb[0];
}

// ---------------- launch ----------------

extern "C" void kernel_launch(void* const* d_in, const int* in_sizes, int n_in,
                              void* d_out, int out_size, void* d_ws, size_t ws_size,
                              hipStream_t stream) {
    (void)in_sizes; (void)n_in; (void)out_size; (void)ws_size;

    const float* x     = (const float*)d_in[0];
    const int*   eidx  = (const int*)d_in[1];
    const int*   batch = (const int*)d_in[2];
    const int*   src   = eidx;
    const int*   dst   = eidx + EE;

    const float *W[4], *asrc[4], *adst[4], *bias[4], *gam[4], *bet[4], *mu[4], *var[4];
    for (int l = 0; l < 4; l++) {
        int base = 3 + 8 * l;
        W[l]    = (const float*)d_in[base + 0];
        asrc[l] = (const float*)d_in[base + 1];
        adst[l] = (const float*)d_in[base + 2];
        bias[l] = (const float*)d_in[base + 3];
        gam[l]  = (const float*)d_in[base + 4];
        bet[l]  = (const float*)d_in[base + 5];
        mu[l]   = (const float*)d_in[base + 6];
        var[l]  = (const float*)d_in[base + 7];
    }
    const float* fcW = (const float*)d_in[35];
    const float* fcb = (const float*)d_in[36];
    float* out = (float*)d_out;

    char* ws = (char*)d_ws;
    size_t off = 0;
    auto alloc = [&](size_t bytes) -> char* {
        char* p = ws + off;
        off += (bytes + 255) & ~(size_t)255;
        return p;
    };
    _Float16*     xh   = (_Float16*)alloc((size_t)NN * HH * sizeof(_Float16));
    _Float16*     wt   = (_Float16*)alloc((size_t)4 * HH * HH * sizeof(_Float16));
    float*        hbuf = (float*)alloc((size_t)NN * HH * sizeof(float));
    float*        agg  = (float*)alloc((size_t)NN * HH * sizeof(float));
    float*        es   = (float*)alloc((size_t)NN * sizeof(float));
    float*        ed   = (float*)alloc((size_t)NN * sizeof(float));
    unsigned int* mM   = (unsigned int*)alloc((size_t)NN * sizeof(unsigned int));
    float*        z    = (float*)alloc((size_t)NN * sizeof(float));
    float*        L    = (float*)alloc((size_t)ENED * sizeof(float));
    float*        EX   = (float*)alloc((size_t)ENED * sizeof(float));
    float*        psum = (float*)alloc((size_t)GG * HH * sizeof(float));
    float*        pcnt = (float*)alloc((size_t)GG * sizeof(float));

    const int T = 256;

    k_cvt_f16<<<(NN * 128 + T - 1) / T, T, 0, stream>>>(x, xh, NN * 128);

    for (int l = 0; l < 4; l++) {
        int K = (l == 0) ? 128 : HH;
        _Float16* wtl = wt + (size_t)l * HH * HH;

        k_wtrans<<<(K * HH + T - 1) / T, T, 0, stream>>>(W[l], wtl, K);

        k_gemm_wmma<<<NN / 32, T, 0, stream>>>(xh, wtl, hbuf, K);

        k_dots<<<(NN * 32 + T - 1) / T, T, 0, stream>>>(hbuf, asrc[l], adst[l], es, ed);

        // mapped(-inf) = ~0xFF800000 = 0x007FFFFF
        k_fill_u32<<<(NN + T - 1) / T, T, 0, stream>>>(mM, 0x007FFFFFu, NN);
        k_fill_f32<<<(NN + T - 1) / T, T, 0, stream>>>(z, 0.f, NN);
        k_fill_f32<<<(NN * HH + T - 1) / T, T, 0, stream>>>(agg, 0.f, NN * HH);

        k_edge_logit_max<<<(ENED + T - 1) / T, T, 0, stream>>>(es, ed, src, dst, L, mM);
        k_edge_exp<<<(ENED + T - 1) / T, T, 0, stream>>>(L, mM, dst, EX, z);
        k_edge_agg<<<(ENED * 32 + T - 1) / T, T, 0, stream>>>(EX, z, src, dst, hbuf, agg);

        k_bn_relu<<<(NN * HH + T - 1) / T, T, 0, stream>>>(agg, bias[l], gam[l], bet[l], mu[l], var[l], xh);
    }

    k_fill_f32<<<(GG * HH + T - 1) / T, T, 0, stream>>>(psum, 0.f, GG * HH);
    k_fill_f32<<<(GG + T - 1) / T, T, 0, stream>>>(pcnt, 0.f, GG);
    k_pool<<<(NN * HH + T - 1) / T, T, 0, stream>>>(agg, batch, psum, pcnt);
    k_fc<<<(GG * 32 + T - 1) / T, T, 0, stream>>>(psum, pcnt, fcW, fcb, out);
}
